// ClusteringLoss_25718264168457
// MI455X (gfx1250) — matrix operand order repaired and checked by reading, not compiled, not run on hardware
//
#include <hip/hip_runtime.h>
#include <hip/hip_bf16.h>
#include <math.h>

typedef __attribute__((ext_vector_type(16))) _Float16 v16h;
typedef __attribute__((ext_vector_type(8)))  float    v8f;
typedef __attribute__((ext_vector_type(4)))  float    v4f;

#define B_ROWS 262144
#define K_CL   1024
#define D_DIM  128
#define EPS_N  1e-6f

// ---------------------------------------------------------------------------
// Zero accumulators (ws dsums + d_out sizes region). Must run every call:
// atomics accumulate into these.
// ---------------------------------------------------------------------------
__global__ void init_zero_kernel(float* __restrict__ dsums,
                                 float* __restrict__ out_sizes) {
    int k = threadIdx.x;           // 1024 threads
    dsums[k]     = 0.0f;
    out_sizes[k] = 0.0f;
}

// ---------------------------------------------------------------------------
// Normalize centroids (f32) -> f16, row-major [K][D].
// One block (128 threads) per centroid.
// ---------------------------------------------------------------------------
__global__ __launch_bounds__(128)
void centroid_norm_kernel(const float* __restrict__ cent,
                          _Float16* __restrict__ centh) {
    __shared__ float red[128];
    const int d = threadIdx.x;
    const int c = blockIdx.x;
    float v = cent[(size_t)c * D_DIM + d];
    red[d] = v * v;
    __syncthreads();
    #pragma unroll
    for (int off = 64; off > 0; off >>= 1) {
        if (d < off) red[d] += red[d + off];
        __syncthreads();
    }
    float rs = 1.0f / (sqrtf(red[0]) + EPS_N);
    centh[(size_t)c * D_DIM + d] = (_Float16)(v * rs);
}

// ---------------------------------------------------------------------------
// 4 chained WMMAs: one 16(M)x16(N) tile over full D=128.
// ---------------------------------------------------------------------------
__device__ __forceinline__ v8f tile_mm(const v16h* __restrict__ af,
                                       v16h b0, v16h b1, v16h b2, v16h b3) {
    v8f c = {};
    c = __builtin_amdgcn_wmma_f32_16x16x32_f16(false, af[0], false, b0,
                                               (short)0, c, false, false);
    c = __builtin_amdgcn_wmma_f32_16x16x32_f16(false, af[1], false, b1,
                                               (short)0, c, false, false);
    c = __builtin_amdgcn_wmma_f32_16x16x32_f16(false, af[2], false, b2,
                                               (short)0, c, false, false);
    c = __builtin_amdgcn_wmma_f32_16x16x32_f16(false, af[3], false, b3,
                                               (short)0, c, false, false);
    return c;
}

// ---------------------------------------------------------------------------
// Load one 16-row A-tile's fragments for this lane:
//   pass 1: sum-of-squares (half-row per lane, joined via shfl_xor 16),
//   pass 2: reload (WGP$-hot), scale by 1/(|row|+eps), pack 4x v16h.
// ---------------------------------------------------------------------------
__device__ __forceinline__ void load_a_tile(const float* __restrict__ frow,
                                            int halfsel, v16h* __restrict__ af) {
    float ssq = 0.0f;
    #pragma unroll
    for (int kb = 0; kb < 4; ++kb) {
        #pragma unroll
        for (int g = 0; g < 2; ++g) {
            const int kbase = kb * 32 + halfsel * 16 + g * 8;
            v4f x0 = *(const v4f*)(frow + kbase);
            v4f x1 = *(const v4f*)(frow + kbase + 4);
            #pragma unroll
            for (int j = 0; j < 4; ++j) ssq += x0[j] * x0[j] + x1[j] * x1[j];
        }
    }
    ssq += __shfl_xor(ssq, 16, 32);      // join complementary half-rows
    const float rs = 1.0f / (sqrtf(ssq) + EPS_N);

    #pragma unroll
    for (int kb = 0; kb < 4; ++kb) {
        #pragma unroll
        for (int g = 0; g < 2; ++g) {
            const int kbase = kb * 32 + halfsel * 16 + g * 8;
            v4f x0 = *(const v4f*)(frow + kbase);
            v4f x1 = *(const v4f*)(frow + kbase + 4);
            #pragma unroll
            for (int j = 0; j < 4; ++j) {
                af[kb][g * 8 + j]     = (_Float16)(x0[j] * rs);
                af[kb][g * 8 + 4 + j] = (_Float16)(x1[j] * rs);
            }
        }
    }
}

// ---------------------------------------------------------------------------
// Main kernel: per wave, 32 rows (two 16-row A sets sharing each B tile ->
// 8 WMMAs per 8 b128 loads); WMMA f32_16x16x32_f16 over all 1024 clusters in
// tiles of 16; running argmax(dot) == argmin(distance).  B tiles use the
// rotation idiom so the scheduler overlaps next-tile loads with the WMMA
// chain (partial s_wait_loadcnt).
//
// A-fragment layout (16-bit A 16x32, same K order as B): lane L in [0,15] holds row
// M=L, K = kb*32 + 0..15 (contiguous); lane L in [16,31] holds row M=L-16,
// K = kb*32 + 16..31.
// B-fragment layout (B 32x16): lane L in [0,15] = column N=L holding
// K = kb*32 + 0..15 (contiguous); lanes 16..31 = column N=L-16 holding
// K = kb*32 + 16..31.
// C/D layout: VGPR r: lanes 0-15 = (M=r, N=lane), lanes 16-31 = (M=r+8, N=lane-16).
// ---------------------------------------------------------------------------
__global__ __launch_bounds__(256)
void cluster_main_kernel(const float* __restrict__ feat,
                         const _Float16* __restrict__ centh,
                         float* __restrict__ out_assign,
                         float* __restrict__ out_sizes,
                         float* __restrict__ dsums) {
    const int lane    = threadIdx.x & 31;
    const int wave    = threadIdx.x >> 5;
    const int halfsel = lane >> 4;      // which half of the K striping
    const int lrow    = lane & 15;      // row-in-tile / column-in-tile
    const int rowBase = blockIdx.x * 256 + wave * 32;

    // ---- build two A-fragment sets (rows rowBase..+15 and rowBase+16..+31)
    v16h afrag0[4], afrag1[4];
    load_a_tile(feat + (size_t)(rowBase + lrow) * D_DIM,      halfsel, afrag0);
    load_a_tile(feat + (size_t)(rowBase + 16 + lrow) * D_DIM, halfsel, afrag1);

    // ---- sweep clusters in tiles of 16
    float bestv0[8], bestv1[8];
    int   besti0[8], besti1[8];
    #pragma unroll
    for (int r = 0; r < 8; ++r) {
        bestv0[r] = -2.0f; besti0[r] = 0;
        bestv1[r] = -2.0f; besti1[r] = 0;
    }

    // lane's B column base: cluster (n0 + lrow), K start = halfsel*16
    const _Float16* cbase = centh + (size_t)lrow * D_DIM + halfsel * 16;

    v16h b0 = *(const v16h*)(cbase);
    v16h b1 = *(const v16h*)(cbase + 32);
    v16h b2 = *(const v16h*)(cbase + 64);
    v16h b3 = *(const v16h*)(cbase + 96);

    for (int n0 = 0; n0 < K_CL; n0 += 16) {
        // issue next tile's loads (wraps to tile 0 on last iter: branch-free,
        // harmless cache hit) so they overlap the WMMA chains below.
        const _Float16* np = cbase + (size_t)((n0 + 16) & (K_CL - 1)) * D_DIM;
        v16h nb0 = *(const v16h*)(np);
        v16h nb1 = *(const v16h*)(np + 32);
        v16h nb2 = *(const v16h*)(np + 64);
        v16h nb3 = *(const v16h*)(np + 96);

        v8f c0 = tile_mm(afrag0, b0, b1, b2, b3);
        v8f c1 = tile_mm(afrag1, b0, b1, b2, b3);

        const int nidx = n0 + lrow;      // this lane's cluster column
        #pragma unroll
        for (int r = 0; r < 8; ++r) {
            if (c0[r] > bestv0[r]) { bestv0[r] = c0[r]; besti0[r] = nidx; }
            if (c1[r] > bestv1[r]) { bestv1[r] = c1[r]; besti1[r] = nidx; }
        }

        b0 = nb0; b1 = nb1; b2 = nb2; b3 = nb3;
    }

    // ---- argmax reduction across the 16 lanes sharing each row
    // (lanes 0-15 reduce rows base+r, lanes 16-31 reduce rows base+8+r)
    #pragma unroll
    for (int s = 0; s < 2; ++s) {
        #pragma unroll
        for (int r = 0; r < 8; ++r) {
            float v = s ? bestv1[r] : bestv0[r];
            int   i = s ? besti1[r] : besti0[r];
            #pragma unroll
            for (int off = 1; off < 16; off <<= 1) {
                float ov = __shfl_xor(v, off, 32);
                int   oi = __shfl_xor(i, off, 32);
                // max dot == min distance; tie-break lowest index (argmin rule)
                if (ov > v || (ov == v && oi < i)) { v = ov; i = oi; }
            }
            if ((lane & 15) == 0) {
                const int   row  = rowBase + s * 16 + halfsel * 8 + r;
                const float mind = 1.0f - v;     // TEMPERATURE == 1
                out_assign[row] = (float)i;
                atomicAdd(&out_sizes[i], 1.0f);
                atomicAdd(&dsums[i], mind);
            }
        }
    }
}

// ---------------------------------------------------------------------------
// Finalize: avg distances per cluster; loss = sum(dsums)/B (since each row's
// min distance is counted exactly once in its assigned cluster's sum).
// ---------------------------------------------------------------------------
__global__ __launch_bounds__(1024)
void finalize_kernel(const float* __restrict__ dsums,
                     const float* __restrict__ sizes,
                     float* __restrict__ out_avg,
                     float* __restrict__ out_loss) {
    __shared__ float red[1024];
    const int k = threadIdx.x;
    const float s  = sizes[k];
    const float ds = dsums[k];
    out_avg[k] = (s > 0.0f) ? ds / fmaxf(s, 1.0f) : 0.0f;
    red[k] = ds;
    __syncthreads();
    #pragma unroll
    for (int off = 512; off > 0; off >>= 1) {
        if (k < off) red[k] += red[k + off];
        __syncthreads();
    }
    if (k == 0) {
        float loss = red[0] / (float)B_ROWS;
        if (isnan(loss) || isinf(loss)) loss = 0.0f;
        *out_loss = loss;
    }
}

// ---------------------------------------------------------------------------
extern "C" void kernel_launch(void* const* d_in, const int* in_sizes, int n_in,
                              void* d_out, int out_size, void* d_ws, size_t ws_size,
                              hipStream_t stream) {
    const float* feat = (const float*)d_in[0];   // [262144, 128] f32
    const float* cent = (const float*)d_in[1];   // [1024, 128]   f32

    float* out        = (float*)d_out;
    float* out_loss   = out;                         // [1]
    float* out_assign = out + 1;                     // [B] (as float)
    float* out_sizes  = out + 1 + B_ROWS;            // [K]
    float* out_avg    = out + 1 + B_ROWS + K_CL;     // [K]

    _Float16* centh = (_Float16*)d_ws;                                   // 256 KB
    float*    dsums = (float*)((char*)d_ws + (size_t)K_CL * D_DIM * 2);  // 4 KB

    init_zero_kernel<<<dim3(1), dim3(1024), 0, stream>>>(dsums, out_sizes);
    centroid_norm_kernel<<<dim3(K_CL), dim3(128), 0, stream>>>(cent, centh);
    cluster_main_kernel<<<dim3(B_ROWS / 256), dim3(256), 0, stream>>>(
        feat, centh, out_assign, out_sizes, dsums);
    finalize_kernel<<<dim3(1), dim3(1024), 0, stream>>>(dsums, out_sizes,
                                                        out_avg, out_loss);
}
